// PointNetFeaturePropagationLite_27547920236660
// MI455X (gfx1250) — matrix operand order repaired and checked by reading, not compiled, not run on hardware
//
#include <hip/hip_runtime.h>

// ---------------------------------------------------------------------------
// Problem constants (from reference setup_inputs)
// ---------------------------------------------------------------------------
#define B_   8
#define N_   8192
#define S_   2048
#define C1_  128
#define C2_  256
#define IN_  384          // C1 + C2
#define OUT_ 256
#define M_   (B_ * N_)    // 65536 rows

typedef __bf16 bf16_t;
typedef bf16_t v16bf __attribute__((ext_vector_type(16)));
typedef bf16_t v8bf  __attribute__((ext_vector_type(8)));
typedef float  v8f   __attribute__((ext_vector_type(8)));

__device__ __forceinline__ bf16_t f2bf(float f) {
  unsigned u = __builtin_bit_cast(unsigned, f);
  u += 0x7FFFu + ((u >> 16) & 1u);              // round-to-nearest-even
  unsigned short h = (unsigned short)(u >> 16);
  return __builtin_bit_cast(bf16_t, h);
}

// A-fragment: two 16-byte chunks at +0 and +8 elements (16 contiguous K values)
__device__ __forceinline__ v16bf load_afrag(const bf16_t* p) {
  v8bf lo = *(const v8bf*)p;
  v8bf hi = *(const v8bf*)(p + 8);
  return __builtin_shufflevector(lo, hi, 0, 1, 2, 3, 4, 5, 6, 7,
                                         8, 9, 10, 11, 12, 13, 14, 15);
}

// B-fragment: 16 contiguous K values of one column (transposed weights)
__device__ __forceinline__ v16bf load_bfrag(const bf16_t* p) {
  v8bf lo = *(const v8bf*)p;
  v8bf hi = *(const v8bf*)(p + 8);
  return __builtin_shufflevector(lo, hi, 0, 1, 2, 3, 4, 5, 6, 7,
                                         8, 9, 10, 11, 12, 13, 14, 15);
}

// ---------------------------------------------------------------------------
// Kernel 1: transpose + convert weights to bf16.
// W1t[n*IN_+k]  = bf16(W1[k*OUT_+n])   (OUT_ x IN_)
// W2t[n*OUT_+k] = bf16(W2[k*OUT_+n])   (OUT_ x OUT_)
// ---------------------------------------------------------------------------
__global__ void wconvert_kernel(const float* __restrict__ W1,
                                const float* __restrict__ W2,
                                bf16_t* __restrict__ W1t,
                                bf16_t* __restrict__ W2t) {
  int idx = blockIdx.x * blockDim.x + threadIdx.x;
  if (idx < OUT_ * IN_) {
    int n = idx / IN_, k = idx % IN_;
    W1t[idx] = f2bf(W1[k * OUT_ + n]);
  }
  if (idx < OUT_ * OUT_) {
    int n = idx / OUT_, k = idx % OUT_;
    W2t[idx] = f2bf(W2[k * OUT_ + n]);
  }
}

// ---------------------------------------------------------------------------
// Kernel 2: per-point 3-NN + inverse-distance interpolation + concat.
// One block (128 threads) per query point; xyz2 batch tile staged in LDS.
// ---------------------------------------------------------------------------
__device__ __forceinline__ void top3_insert(float d, int i,
                                            float& d0, int& i0,
                                            float& d1, int& i1,
                                            float& d2, int& i2) {
  if (d < d2) {
    if (d < d0)      { d2 = d1; i2 = i1; d1 = d0; i1 = i0; d0 = d; i0 = i; }
    else if (d < d1) { d2 = d1; i2 = i1; d1 = d;  i1 = i; }
    else             { d2 = d;  i2 = i; }
  }
}

__global__ __launch_bounds__(128) void interp_concat_kernel(
    const float* __restrict__ xyz1, const float* __restrict__ xyz2,
    const float* __restrict__ feat1, const float* __restrict__ feat2,
    bf16_t* __restrict__ X) {
  __shared__ float s_xyz[S_ * 3];     // 24 KB
  __shared__ float s_d[3 * 128];
  __shared__ int   s_i[3 * 128];
  __shared__ float s_w[3];
  __shared__ int   s_n[3];

  const int t = threadIdx.x;
  const int p = blockIdx.x;           // global point index in [0, M_)
  const int b = p / N_;

  const float* x2 = xyz2 + (size_t)b * S_ * 3;
  for (int i = t; i < S_ * 3; i += 128) s_xyz[i] = x2[i];
  __syncthreads();

  const float px = xyz1[(size_t)p * 3 + 0];
  const float py = xyz1[(size_t)p * 3 + 1];
  const float pz = xyz1[(size_t)p * 3 + 2];

  float d0 = 1e30f, d1 = 1e30f, d2 = 1e30f;
  int   i0 = 0,     i1 = 0,     i2 = 0;
  for (int s = t; s < S_; s += 128) {
    float dx = px - s_xyz[s * 3 + 0];
    float dy = py - s_xyz[s * 3 + 1];
    float dz = pz - s_xyz[s * 3 + 2];
    float d = dx * dx + dy * dy + dz * dz;
    top3_insert(d, s, d0, i0, d1, i1, d2, i2);
  }

  s_d[t] = d0; s_d[128 + t] = d1; s_d[256 + t] = d2;
  s_i[t] = i0; s_i[128 + t] = i1; s_i[256 + t] = i2;
  __syncthreads();

  for (int off = 64; off > 0; off >>= 1) {
    if (t < off) {
      float a0 = s_d[t], a1 = s_d[128 + t], a2 = s_d[256 + t];
      int   j0 = s_i[t], j1 = s_i[128 + t], j2 = s_i[256 + t];
#pragma unroll
      for (int q = 0; q < 3; ++q)
        top3_insert(s_d[q * 128 + t + off], s_i[q * 128 + t + off],
                    a0, j0, a1, j1, a2, j2);
      s_d[t] = a0; s_d[128 + t] = a1; s_d[256 + t] = a2;
      s_i[t] = j0; s_i[128 + t] = j1; s_i[256 + t] = j2;
    }
    __syncthreads();
  }

  if (t == 0) {
    float w0 = 1.0f / fmaxf(sqrtf(fmaxf(s_d[0],   0.0f)), 1e-10f);
    float w1 = 1.0f / fmaxf(sqrtf(fmaxf(s_d[128], 0.0f)), 1e-10f);
    float w2 = 1.0f / fmaxf(sqrtf(fmaxf(s_d[256], 0.0f)), 1e-10f);
    float ws = w0 + w1 + w2;
    s_w[0] = w0 / ws; s_w[1] = w1 / ws; s_w[2] = w2 / ws;
    s_n[0] = s_i[0];  s_n[1] = s_i[128]; s_n[2] = s_i[256];
  }
  __syncthreads();

  bf16_t* xr = X + (size_t)p * IN_;
  const float* f1 = feat1 + (size_t)p * C1_;
  xr[t] = f2bf(f1[t]);

  const float w0 = s_w[0], w1 = s_w[1], w2 = s_w[2];
  const size_t fb = (size_t)b * S_ * C2_;
  const float* r0 = feat2 + fb + (size_t)s_n[0] * C2_;
  const float* r1 = feat2 + fb + (size_t)s_n[1] * C2_;
  const float* r2 = feat2 + fb + (size_t)s_n[2] * C2_;
  for (int c = t; c < C2_; c += 128) {
    float v = w0 * r0[c] + w1 * r1[c] + w2 * r2[c];
    xr[C1_ + c] = f2bf(v);
  }
}

// ---------------------------------------------------------------------------
// Kernel 3/4: software-pipelined WMMA bf16 GEMM + bias + relu.
//   A : M_ x K row-major bf16
//   Wt: OUT_ x K row-major bf16 (transposed weights)
// Block = 256 threads = 8 waves arranged 2(M) x 4(N).
// Block tile = 32 (M) x 256 (N); wave tile = 16 x 64 (4 WMMA accumulators).
// K-loop is double-buffered: fragments for step k+1 are loaded before the
// WMMAs of step k issue, so global loads overlap matrix ops.
// ---------------------------------------------------------------------------
template <int K, bool STORE_F32>
__global__ __launch_bounds__(256) void gemm_bias_relu_kernel(
    const bf16_t* __restrict__ A, const bf16_t* __restrict__ Wt,
    const float* __restrict__ bias,
    bf16_t* __restrict__ Obf, float* __restrict__ Of) {
  const int lane  = threadIdx.x & 31;
  const int wave  = threadIdx.x >> 5;
  const int half  = lane >> 4;     // 0 or 1
  const int l15   = lane & 15;
  const int waveN = wave & 3;      // 4 waves across N
  const int waveM = wave >> 2;     // 2 waves across M

  const int m0 = blockIdx.x * 32 + waveM * 16;
  const int n0 = waveN * 64;

  const bf16_t* arow = A + (size_t)(m0 + l15) * K + half * 16;
  const bf16_t* br[4];
#pragma unroll
  for (int j = 0; j < 4; ++j)
    br[j] = Wt + (size_t)(n0 + j * 16 + l15) * K + half * 16;

  v8f acc[4] = {{}, {}, {}, {}};

  // prologue: fragments for kk = 0
  v16bf af = load_afrag(arow);
  v16bf bfg[4];
#pragma unroll
  for (int j = 0; j < 4; ++j) bfg[j] = load_bfrag(br[j]);

  for (int kk = 32; kk < K; kk += 32) {
    __builtin_prefetch(arow + kk + 32, 0, 1);   // global_prefetch_b8
    // issue next-step loads before consuming current fragments
    v16bf afn = load_afrag(arow + kk);
    v16bf bfn0 = load_bfrag(br[0] + kk);
    v16bf bfn1 = load_bfrag(br[1] + kk);
    v16bf bfn2 = load_bfrag(br[2] + kk);
    v16bf bfn3 = load_bfrag(br[3] + kk);

    acc[0] = __builtin_amdgcn_wmma_f32_16x16x32_bf16(false, af, false, bfg[0],
                                                     (short)0, acc[0], false, false);
    acc[1] = __builtin_amdgcn_wmma_f32_16x16x32_bf16(false, af, false, bfg[1],
                                                     (short)0, acc[1], false, false);
    acc[2] = __builtin_amdgcn_wmma_f32_16x16x32_bf16(false, af, false, bfg[2],
                                                     (short)0, acc[2], false, false);
    acc[3] = __builtin_amdgcn_wmma_f32_16x16x32_bf16(false, af, false, bfg[3],
                                                     (short)0, acc[3], false, false);

    af = afn;
    bfg[0] = bfn0; bfg[1] = bfn1; bfg[2] = bfn2; bfg[3] = bfn3;
  }

  // epilogue of pipeline: last K-step
  acc[0] = __builtin_amdgcn_wmma_f32_16x16x32_bf16(false, af, false, bfg[0],
                                                   (short)0, acc[0], false, false);
  acc[1] = __builtin_amdgcn_wmma_f32_16x16x32_bf16(false, af, false, bfg[1],
                                                   (short)0, acc[1], false, false);
  acc[2] = __builtin_amdgcn_wmma_f32_16x16x32_bf16(false, af, false, bfg[2],
                                                   (short)0, acc[2], false, false);
  acc[3] = __builtin_amdgcn_wmma_f32_16x16x32_bf16(false, af, false, bfg[3],
                                                   (short)0, acc[3], false, false);

  // bias + relu + store.  C/D layout: VGPR r, lane l -> M = r + 8*(l>=16), N = l%16
#pragma unroll
  for (int j = 0; j < 4; ++j) {
    const int n = n0 + j * 16 + l15;
    const float bv = bias[n];
#pragma unroll
    for (int r = 0; r < 8; ++r) {
      const int m = m0 + r + half * 8;
      float v = fmaxf(acc[j][r] + bv, 0.0f);
      if (STORE_F32) Of[(size_t)m * OUT_ + n] = v;
      else           Obf[(size_t)m * OUT_ + n] = f2bf(v);
    }
  }
}

// ---------------------------------------------------------------------------
// Launcher
// ---------------------------------------------------------------------------
extern "C" void kernel_launch(void* const* d_in, const int* in_sizes, int n_in,
                              void* d_out, int out_size, void* d_ws, size_t ws_size,
                              hipStream_t stream) {
  const float* xyz1  = (const float*)d_in[0];
  const float* xyz2  = (const float*)d_in[1];
  const float* feat1 = (const float*)d_in[2];
  const float* feat2 = (const float*)d_in[3];
  const float* W1    = (const float*)d_in[4];
  const float* b1    = (const float*)d_in[5];
  const float* W2    = (const float*)d_in[6];
  const float* b2    = (const float*)d_in[7];
  float* out = (float*)d_out;

  // workspace layout (bf16): X (M x IN), H (M x OUT), W1t (OUT x IN), W2t (OUT x OUT)
  char* ws = (char*)d_ws;
  bf16_t* X   = (bf16_t*)ws;
  bf16_t* H   = (bf16_t*)(ws + (size_t)M_ * IN_ * sizeof(bf16_t));
  bf16_t* W1t = (bf16_t*)(ws + (size_t)M_ * IN_ * sizeof(bf16_t)
                             + (size_t)M_ * OUT_ * sizeof(bf16_t));
  bf16_t* W2t = W1t + (size_t)OUT_ * IN_;

  wconvert_kernel<<<(OUT_ * IN_ + 255) / 256, 256, 0, stream>>>(W1, W2, W1t, W2t);

  interp_concat_kernel<<<M_, 128, 0, stream>>>(xyz1, xyz2, feat1, feat2, X);

  gemm_bias_relu_kernel<IN_, false><<<M_ / 32, 256, 0, stream>>>(
      X, W1t, b1, H, nullptr);

  gemm_bias_relu_kernel<OUT_, true><<<M_ / 32, 256, 0, stream>>>(
      H, W2t, b2, nullptr, out);
}
